// FinalModel_26448408609305
// MI455X (gfx1250) — matrix operand order repaired and checked by reading, not compiled, not run on hardware
//
#include <hip/hip_runtime.h>
#include <hip/hip_bf16.h>
#include <math.h>

// ---------------------------------------------------------------------------
// PaiNN-style GNN forward on gfx1250 (MI455X).
// All GEMMs via v_wmma_f32_16x16x32_f16 (f16 in, f32 accum).
// Edge phase fully fused (async-gather -> phi1 -> silu -> phi2 & rbf-GEMM ->
// combine -> atomic scatter); 32 edges/block so every weight B-fragment
// feeds two WMMA M-tiles (halves L2 weight traffic vs 16 edges/block).
// ---------------------------------------------------------------------------

#define N_NODES 10000
#define E_EDGES 160000
#define HCH     128
#define NRBF    20
#define KRBF    32      // rbf K padded to 32 for WMMA
#define NGRAPH  64
#define NLAYER  4
#define TILE_E  32      // edges per block in the fused edge kernel
#define CUTOFF_F 5.0f
#define PI_F 3.14159265358979f

typedef __attribute__((ext_vector_type(16))) _Float16 v16h;
typedef __attribute__((ext_vector_type(8)))  _Float16 v8h;
typedef __attribute__((ext_vector_type(8)))  float    v8f;

#if defined(__has_builtin)
#  if __has_builtin(__builtin_amdgcn_global_load_async_to_lds_b128)
#    define HAVE_ASYNC_LDS 1
#  endif
#endif
#ifndef HAVE_ASYNC_LDS
#  define HAVE_ASYNC_LDS 0
#endif

#if HAVE_ASYNC_LDS
// builtin signature (from hipcc diagnostic): param0 = <4 x i32> addrspace(1)*,
// then lds ptr, imm offset, imm cpol.
typedef int v4i_ __attribute__((vector_size(16)));
typedef __attribute__((address_space(1))) v4i_* gptr4;
typedef __attribute__((address_space(3))) v4i_* lptr4;
__device__ __forceinline__ void async_ld128(const void* g, void* l) {
  __builtin_amdgcn_global_load_async_to_lds_b128((gptr4)g, (lptr4)l, 0, 0);
}
__device__ __forceinline__ void async_wait0() {
#if __has_builtin(__builtin_amdgcn_s_wait_asynccnt)
  __builtin_amdgcn_s_wait_asynccnt(0);
#else
  asm volatile("s_wait_asynccnt 0" ::: "memory");
#endif
}
#endif

// ---------------- WMMA fragment helpers (k = 16*(lane>>4) + s in both A and B) --

__device__ __forceinline__ v8f wmma_f16(v16h a, v16h b, v8f c) {
  return __builtin_amdgcn_wmma_f32_16x16x32_f16(
      /*neg_a=*/false, a, /*neg_b=*/false, b,
      /*c_mod=*/(short)0, c, /*reuse_a=*/false, /*reuse_b=*/false);
}

// A fragment: 16x32 tile, row-major source with leading dim ld (halfs).
__device__ __forceinline__ v16h ldA(const _Float16* base, int ld, int k0, int lane) {
  int m  = lane & 15;
  int hi = lane >> 4;
  const _Float16* p = base + (size_t)m * ld + k0;
  return *reinterpret_cast<const v16h*>(p + hi * 16);
}

// B fragment: K x N tile taken from weight stored (N x Kpad) row-major.
__device__ __forceinline__ v16h ldB(const _Float16* W, int ldw, int n0, int k0, int lane) {
  int n    = lane & 15;
  int koff = (lane >> 4) * 16;
  const _Float16* p = W + (size_t)(n0 + n) * ldw + k0 + koff;
  return *reinterpret_cast<const v16h*>(p);
}

__device__ __forceinline__ float fast_rcp(float x) {
#if defined(__has_builtin) && __has_builtin(__builtin_amdgcn_rcpf)
  return __builtin_amdgcn_rcpf(x);
#else
  return 1.f / x;
#endif
}
__device__ __forceinline__ float sigmoid_f(float x) { return fast_rcp(1.f + __expf(-x)); }
__device__ __forceinline__ float silu_f(float x)    { return x * sigmoid_f(x); }

// ---------------- small utility kernels -------------------------------------

__global__ void cvt_w16_kernel(const float* __restrict__ W, _Float16* __restrict__ out,
                               int rows, int cols, int colsPad) {
  int i = blockIdx.x * blockDim.x + threadIdx.x;
  int total = rows * colsPad;
  if (i >= total) return;
  int r = i / colsPad, c = i - r * colsPad;
  out[i] = (c < cols) ? (_Float16)W[r * cols + c] : (_Float16)0.f;
}

__global__ void cvt_f16_kernel(const float* __restrict__ x, _Float16* __restrict__ y, int n) {
  int i = blockIdx.x * blockDim.x + threadIdx.x;
  if (i < n) y[i] = (_Float16)x[i];
}

__global__ void copy_f32_kernel(const float* __restrict__ x, float* __restrict__ y, int n) {
  int i = blockIdx.x * blockDim.x + threadIdx.x;
  if (i < n) y[i] = x[i];
}

__global__ void zero_f32_kernel(float* __restrict__ x, int n) {
  int i = blockIdx.x * blockDim.x + threadIdx.x;
  if (i < n) x[i] = 0.f;
}

__global__ void init_nodes_kernel(const int* __restrict__ z, const float* __restrict__ embed,
                                  float* __restrict__ s, float* __restrict__ v,
                                  int nhc, int total3) {
  int i = blockIdx.x * blockDim.x + threadIdx.x;
  if (i >= total3) return;
  v[i] = 0.f;
  if (i < nhc) {
    int n = i >> 7, h = i & 127;
    s[i] = embed[z[n] * HCH + h];
  }
}

__global__ void edge_geom_kernel(const float* __restrict__ pos, const int* __restrict__ eidx,
                                 float* __restrict__ ea, _Float16* __restrict__ rbf16, int E) {
  int e = blockIdx.x * blockDim.x + threadIdx.x;
  if (e >= E) return;
  int r = eidx[e], c = eidx[E + e];
  float dx = pos[r * 3 + 0] - pos[c * 3 + 0];
  float dy = pos[r * 3 + 1] - pos[c * 3 + 1];
  float dz = pos[r * 3 + 2] - pos[c * 3 + 2];
  float d = sqrtf(dx * dx + dy * dy + dz * dz + 1e-12f);
  ea[e * 3 + 0] = dx; ea[e * 3 + 1] = dy; ea[e * 3 + 2] = dz;
  float cut  = 0.5f * (__cosf(PI_F * d / CUTOFF_F) + 1.f);
  float mask = (d < CUTOFF_F) ? 1.f : 0.f;
  float scale = fast_rcp(d + 1e-15f) * cut * mask;
#pragma unroll
  for (int k = 0; k < KRBF; ++k) {
    float val = 0.f;
    if (k < NRBF) val = __sinf(d * (float)(k + 1) * PI_F / CUTOFF_F) * scale;
    rbf16[(size_t)e * KRBF + k] = (_Float16)val;
  }
}

// build sv16 = [s | ||v||] as f16 (N x 256)
__global__ void vnorm_sv_kernel(const float* __restrict__ s, const float* __restrict__ v,
                                _Float16* __restrict__ sv16, int Nn) {
  int i = blockIdx.x * blockDim.x + threadIdx.x;
  if (i >= Nn * HCH) return;
  int n = i >> 7, h = i & 127;
  float v0 = v[((size_t)0 * Nn + n) * HCH + h];
  float v1 = v[((size_t)1 * Nn + n) * HCH + h];
  float v2 = v[((size_t)2 * Nn + n) * HCH + h];
  float vn = sqrtf(v0 * v0 + v1 * v1 + v2 * v2 + 1e-12f);
  sv16[(size_t)n * 256 + h]       = (_Float16)s[i];
  sv16[(size_t)n * 256 + 128 + h] = (_Float16)vn;
}

// ---------------- fused edge message kernel ---------------------------------
// TILE_E edges per 256-thread block (8 wave32s, 2 M-tiles per wave).
// Wave w owns channel slice [16w,16w+16): phi1 tile w, phi2 & rbf tiles
// {w, w+8, w+16}, then combine + atomic scatter into s / v.
__global__ void __launch_bounds__(256) edge_kernel(
    const int* __restrict__ rowIdx, const int* __restrict__ colIdx,
    const _Float16* __restrict__ s16, const float* __restrict__ v_old,
    const _Float16* __restrict__ rbf16, const float* __restrict__ ea,
    const _Float16* __restrict__ w1, const float* __restrict__ b1,
    const _Float16* __restrict__ w2, const float* __restrict__ b2,
    const _Float16* __restrict__ wr,
    const float* __restrict__ bws, const float* __restrict__ bwv,
    const float* __restrict__ bwvs,
    float* __restrict__ sAcc, float* __restrict__ vAcc) {
  __shared__ alignas(64) _Float16 sA[TILE_E * HCH];   // 8 KB
  __shared__ alignas(64) _Float16 t1[TILE_E * HCH];   // 8 KB
  __shared__ alignas(64) _Float16 rA[TILE_E * KRBF];  // 2 KB
  __shared__ int   rowL[TILE_E];
  __shared__ int   colL[TILE_E];
  __shared__ float eaL[TILE_E][3];

  int tid = threadIdx.x;
  int e0  = blockIdx.x * TILE_E;

  // gather s16[row] rows (TILE_E x 256B) and rbf tiles (TILE_E x 64B)
#if HAVE_ASYNC_LDS
#pragma unroll
  for (int i = tid; i < TILE_E * 16; i += 256) {
    int el = i >> 4, ck = i & 15;
    int r = rowIdx[e0 + el];
    async_ld128(&s16[(size_t)r * HCH + ck * 8], &sA[el * HCH + ck * 8]);
  }
  if (tid < TILE_E * 4) {
    int el = tid >> 2, ck = tid & 3;
    async_ld128(&rbf16[(size_t)(e0 + el) * KRBF + ck * 8], &rA[el * KRBF + ck * 8]);
  }
#else
#pragma unroll
  for (int i = tid; i < TILE_E * 16; i += 256) {
    int el = i >> 4, ck = i & 15;
    int r = rowIdx[e0 + el];
    *reinterpret_cast<uint4*>(&sA[el * HCH + ck * 8]) =
        *reinterpret_cast<const uint4*>(&s16[(size_t)r * HCH + ck * 8]);
  }
  if (tid < TILE_E * 4) {
    int el = tid >> 2, ck = tid & 3;
    *reinterpret_cast<uint4*>(&rA[el * KRBF + ck * 8]) =
        *reinterpret_cast<const uint4*>(&rbf16[(size_t)(e0 + el) * KRBF + ck * 8]);
  }
#endif
  if (tid < TILE_E) {
    int e = e0 + tid;
    rowL[tid] = rowIdx[e];
    colL[tid] = colIdx[e];
    eaL[tid][0] = ea[e * 3 + 0];
    eaL[tid][1] = ea[e * 3 + 1];
    eaL[tid][2] = ea[e * 3 + 2];
  }
#if HAVE_ASYNC_LDS
  async_wait0();
#endif
  __syncthreads();

  int wv = tid >> 5, lane = tid & 31;
  int n  = lane & 15, hi = lane >> 4;
  const v8f vzero = {0.f, 0.f, 0.f, 0.f, 0.f, 0.f, 0.f, 0.f};

  // -- phi1 + silu -> t1 (f16, in LDS); one B-fragment feeds two M-tiles --
  {
    __builtin_prefetch(w1 + (size_t)(wv * 16) * HCH, 0, 3);
    v8f acc[2] = {vzero, vzero};
#pragma unroll
    for (int k0 = 0; k0 < HCH; k0 += 32) {
      v16h b = ldB(w1, HCH, wv * 16, k0, lane);
      acc[0] = wmma_f16(ldA(sA, HCH, k0, lane), b, acc[0]);
      acc[1] = wmma_f16(ldA(sA + 16 * HCH, HCH, k0, lane), b, acc[1]);
    }
    int nch = wv * 16 + n;
    float bb = b1[nch];
#pragma unroll
    for (int mt = 0; mt < 2; ++mt)
#pragma unroll
      for (int r8 = 0; r8 < 8; ++r8) {
        float x = acc[mt][r8] + bb;
        t1[(mt * 16 + hi * 8 + r8) * HCH + nch] = (_Float16)silu_f(x);
      }
  }
  __syncthreads();

  // -- rbf GEMM (K=32, one WMMA per slice per M-tile) --
  int ch0 = wv * 16;
  v8f rs[2] = {vzero, vzero}, rv[2] = {vzero, vzero}, rvs[2] = {vzero, vzero};
  {
    v16h ar0 = ldA(rA, KRBF, 0, lane);
    v16h ar1 = ldA(rA + 16 * KRBF, KRBF, 0, lane);
    v16h bS  = ldB(wr, KRBF, ch0,           0, lane);
    v16h bV  = ldB(wr, KRBF, HCH + ch0,     0, lane);
    v16h bVS = ldB(wr, KRBF, 2 * HCH + ch0, 0, lane);
    rs[0]  = wmma_f16(ar0, bS,  rs[0]);   rs[1]  = wmma_f16(ar1, bS,  rs[1]);
    rv[0]  = wmma_f16(ar0, bV,  rv[0]);   rv[1]  = wmma_f16(ar1, bV,  rv[1]);
    rvs[0] = wmma_f16(ar0, bVS, rvs[0]);  rvs[1] = wmma_f16(ar1, bVS, rvs[1]);
  }

  // -- phi2 (three channel-aligned slices x two M-tiles) --
  v8f hs[2] = {vzero, vzero}, hv[2] = {vzero, vzero}, hvs[2] = {vzero, vzero};
  {
    __builtin_prefetch(w2 + (size_t)ch0 * HCH, 0, 3);
#pragma unroll
    for (int k0 = 0; k0 < HCH; k0 += 32) {
      v16h a0 = ldA(t1, HCH, k0, lane);
      v16h a1 = ldA(t1 + 16 * HCH, HCH, k0, lane);
      v16h bS = ldB(w2, HCH, ch0, k0, lane);
      hs[0] = wmma_f16(a0, bS, hs[0]);  hs[1] = wmma_f16(a1, bS, hs[1]);
      v16h bV = ldB(w2, HCH, HCH + ch0, k0, lane);
      hv[0] = wmma_f16(a0, bV, hv[0]);  hv[1] = wmma_f16(a1, bV, hv[1]);
      v16h bVS = ldB(w2, HCH, 2 * HCH + ch0, k0, lane);
      hvs[0] = wmma_f16(a0, bVS, hvs[0]);  hvs[1] = wmma_f16(a1, bVS, hvs[1]);
    }
  }

  // -- combine + atomic scatter --
  int ch = ch0 + n;
  float b2s = b2[ch], b2v = b2[HCH + ch], b2vs = b2[2 * HCH + ch];
  float bs_ = bws[ch], bv_ = bwv[ch], bvs_ = bwvs[ch];
#pragma unroll
  for (int mt = 0; mt < 2; ++mt) {
#pragma unroll
    for (int r8 = 0; r8 < 8; ++r8) {
      int el = mt * 16 + hi * 8 + r8;
      float dsv = (hs[mt][r8]  + b2s)  * (rs[mt][r8]  + bs_);
      float cv  = (hv[mt][r8]  + b2v)  * (rv[mt][r8]  + bv_);
      float cvs = (hvs[mt][r8] + b2vs) * (rvs[mt][r8] + bvs_);
      int c = colL[el], rr = rowL[el];
      atomicAdd(&sAcc[(size_t)c * HCH + ch], dsv);
#pragma unroll
      for (int d = 0; d < 3; ++d) {
        float dv = v_old[((size_t)d * N_NODES + rr) * HCH + ch] * cv +
                   eaL[el][d] * cvs;
        atomicAdd(&vAcc[((size_t)d * N_NODES + c) * HCH + ch], dv);
      }
    }
  }
}

// ---------------- generic WMMA GEMM: C = act(A @ W^T + bias) -----------------
// A: (M x K) f16 row-major lda; W: (N x Kpad) f16 row-major ldw.
// Two 16x16 M-tiles per wave (shared B fragments); 8 waves tile along N.
__global__ void __launch_bounds__(256) gemm_wmma_kernel(
    const _Float16* __restrict__ A, int lda,
    const _Float16* __restrict__ W, int ldw,
    const float* __restrict__ bias,
    float* __restrict__ Cf, _Float16* __restrict__ Ch, int ldc,
    int M, int Nn, int K, int act) {
  int wv = threadIdx.x >> 5, lane = threadIdx.x & 31;
  int mt0 = blockIdx.x * 2;
  int nt  = blockIdx.y * 8 + wv;
  if (mt0 * 16 >= M || nt * 16 >= Nn) return;
  bool hasM1 = ((mt0 + 1) * 16) < M;

  const _Float16* Ab0 = A + (size_t)mt0 * 16 * lda;
  const _Float16* Ab1 = Ab0 + (size_t)16 * lda;
  __builtin_prefetch(W + (size_t)(nt * 16) * ldw, 0, 3);
  const v8f vzero = {0.f, 0.f, 0.f, 0.f, 0.f, 0.f, 0.f, 0.f};
  v8f acc0 = vzero, acc1 = vzero;
  for (int k0 = 0; k0 < K; k0 += 32) {
    v16h b = ldB(W, ldw, nt * 16, k0, lane);
    acc0 = wmma_f16(ldA(Ab0, lda, k0, lane), b, acc0);
    if (hasM1) acc1 = wmma_f16(ldA(Ab1, lda, k0, lane), b, acc1);
  }
  int n = lane & 15, hi = lane >> 4;
  int nc = nt * 16 + n;
  float bb = bias ? bias[nc] : 0.f;
#pragma unroll
  for (int r8 = 0; r8 < 8; ++r8) {
    int m = mt0 * 16 + hi * 8 + r8;
    float x = acc0[r8] + bb;
    if (act == 1) x = silu_f(x);
    if (Cf) Cf[(size_t)m * ldc + nc] = x;
    if (Ch) Ch[(size_t)m * ldc + nc] = (_Float16)x;
  }
  if (hasM1) {
#pragma unroll
    for (int r8 = 0; r8 < 8; ++r8) {
      int m = (mt0 + 1) * 16 + hi * 8 + r8;
      float x = acc1[r8] + bb;
      if (act == 1) x = silu_f(x);
      if (Cf) Cf[(size_t)m * ldc + nc] = x;
      if (Ch) Ch[(size_t)m * ldc + nc] = (_Float16)x;
    }
  }
}

// ---------------- gate / layernorm node update ------------------------------
__global__ void __launch_bounds__(128) gate_apply_kernel(
    const float* __restrict__ gpre, const float* __restrict__ gs,
    const float* __restrict__ lnG, const float* __restrict__ lnB,
    const float* __restrict__ wvout, float* __restrict__ s,
    float* __restrict__ v, int Nn) {
  int n = blockIdx.x, h = threadIdx.x;
  __shared__ float red[128];
  float x = gs[(size_t)n * HCH + h];
  red[h] = x;
  __syncthreads();
  for (int st = 64; st > 0; st >>= 1) {
    if (h < st) red[h] += red[h + st];
    __syncthreads();
  }
  float mean = red[0] * (1.f / 128.f);
  __syncthreads();
  float dx = x - mean;
  red[h] = dx * dx;
  __syncthreads();
  for (int st = 64; st > 0; st >>= 1) {
    if (h < st) red[h] += red[h + st];
    __syncthreads();
  }
  float var = red[0] * (1.f / 128.f);
  float ln = dx * rsqrtf(var + 1e-5f) * lnG[h] + lnB[h];
  float gate = sigmoid_f(gpre[(size_t)n * HCH + h]);
  s[(size_t)n * HCH + h] += gate * silu_f(ln);
#pragma unroll
  for (int d = 0; d < 3; ++d) {
    size_t idx = ((size_t)d * Nn + n) * HCH + h;
    v[idx] += gate * wvout[idx];
  }
}

// ---------------- UV . VV aggregation ---------------------------------------
__global__ void aggregate_kernel(const float* __restrict__ aout,
                                 const float* __restrict__ uv,
                                 const float* __restrict__ vvb,
                                 float* __restrict__ s, float* __restrict__ v, int Nn) {
  int i = blockIdx.x * blockDim.x + threadIdx.x;
  if (i >= Nn * HCH) return;
  int n = i >> 7, h = i & 127;
  float dot = 0.f;
#pragma unroll
  for (int d = 0; d < 3; ++d) {
    size_t idx = ((size_t)d * Nn + n) * HCH + h;
    dot += uv[idx] * vvb[idx];
  }
  s[i] += aout[(size_t)n * 384 + h] + aout[(size_t)n * 384 + 128 + h] * dot;
  float avv = aout[(size_t)n * 384 + 256 + h];
#pragma unroll
  for (int d = 0; d < 3; ++d) {
    size_t idx = ((size_t)d * Nn + n) * HCH + h;
    v[idx] += avv * uv[idx];
  }
}

__global__ void pool_kernel(const float* __restrict__ s, const int* __restrict__ batch,
                            float* __restrict__ sg, int Nn) {
  int i = blockIdx.x * blockDim.x + threadIdx.x;
  if (i >= Nn * HCH) return;
  int n = i >> 7, h = i & 127;
  atomicAdd(&sg[(size_t)batch[n] * HCH + h], s[i]);
}

__global__ void __launch_bounds__(128) out2_kernel(const _Float16* __restrict__ h1,
                                                   const float* __restrict__ W2,
                                                   const float* __restrict__ b2,
                                                   float* __restrict__ out) {
  int g = blockIdx.x, h = threadIdx.x;
  __shared__ float red[128];
  red[h] = (float)h1[(size_t)g * HCH + h] * W2[h];
  __syncthreads();
  for (int st = 64; st > 0; st >>= 1) {
    if (h < st) red[h] += red[h + st];
    __syncthreads();
  }
  if (h == 0) out[g] = red[0] + b2[0];
}

// ---------------------------------------------------------------------------
// Host orchestration.
// Input order (jax pytree flatten, sorted dict keys inside params):
//  0:z 1:pos 2:batch 3:edge_index 4:embed
//  5 + 26*l + { 0:U.W 1:U.b 2:V.W 3:V.b 4:Ws.W 5:Ws.b 6:Wv.W 7:Wv.b
//               8:Wvs.W 9:Wvs.b 10:a1.W 11:a1.b 12:a2.W 13:a2.b
//              14:gWg.W 15:gWg.b 16:gWs.W 17:gWs.b 18:gWv.W 19:gWv.b
//              20:ln.g 21:ln.b 22:phi1.W 23:phi1.b 24:phi2.W 25:phi2.b }
//  109:out1.W 110:out1.b 111:out2.W 112:out2.b
// ---------------------------------------------------------------------------
extern "C" void kernel_launch(void* const* d_in, const int* in_sizes, int n_in,
                              void* d_out, int out_size, void* d_ws, size_t ws_size,
                              hipStream_t stream) {
  (void)in_sizes; (void)n_in; (void)out_size; (void)ws_size;
  const int*   z     = (const int*)d_in[0];
  const float* pos   = (const float*)d_in[1];
  const int*   batch = (const int*)d_in[2];
  const int*   eidx  = (const int*)d_in[3];
  const float* embed = (const float*)d_in[4];
  auto PW = [&](int i) { return (const float*)d_in[i]; };
  float* out = (float*)d_out;

  // --- workspace carve-out ---
  char* base = (char*)d_ws;
  size_t off = 0;
  auto alloc = [&](size_t bytes) -> char* {
    off = (off + 255) & ~(size_t)255;
    char* p = base + off;
    off += bytes;
    return p;
  };
  const size_t NHC = (size_t)N_NODES * HCH;
  float*    s     = (float*)alloc(NHC * 4);
  float*    v     = (float*)alloc(3 * NHC * 4);
  float*    v_old = (float*)alloc(3 * NHC * 4);
  _Float16* s16   = (_Float16*)alloc(NHC * 2);
  _Float16* v16   = (_Float16*)alloc(3 * NHC * 2);
  _Float16* sv16  = (_Float16*)alloc((size_t)N_NODES * 256 * 2);
  _Float16* rbf16 = (_Float16*)alloc((size_t)E_EDGES * KRBF * 2);
  float*    ea    = (float*)alloc((size_t)E_EDGES * 3 * 4);
  _Float16* t16   = (_Float16*)alloc(NHC * 2);
  float*    gpre  = (float*)alloc(NHC * 4);
  float*    gs    = (float*)alloc(NHC * 4);
  float*    wvout = (float*)alloc(3 * NHC * 4);   // gWv out, then reused as Uv
  float*    vvout = (float*)alloc(3 * NHC * 4);   // Vv
  float*    aout  = (float*)alloc((size_t)N_NODES * 384 * 4);
  float*    sg    = (float*)alloc((size_t)NGRAPH * HCH * 4);
  _Float16* sg16  = (_Float16*)alloc((size_t)NGRAPH * HCH * 2);
  _Float16* h1_16 = (_Float16*)alloc((size_t)NGRAPH * HCH * 2);
  _Float16* w_out1 = (_Float16*)alloc((size_t)HCH * HCH * 2);
  _Float16 *w_phi1[NLAYER], *w_phi2[NLAYER], *w_rbf[NLAYER], *w_gwg[NLAYER],
           *w_gws[NLAYER], *w_gwv[NLAYER], *w_u[NLAYER], *w_v[NLAYER],
           *w_a1[NLAYER], *w_a2[NLAYER];
  for (int l = 0; l < NLAYER; ++l) {
    w_phi1[l] = (_Float16*)alloc((size_t)HCH * HCH * 2);
    w_phi2[l] = (_Float16*)alloc((size_t)384 * HCH * 2);
    w_rbf[l]  = (_Float16*)alloc((size_t)384 * KRBF * 2);
    w_gwg[l]  = (_Float16*)alloc((size_t)HCH * 256 * 2);
    w_gws[l]  = (_Float16*)alloc((size_t)HCH * 256 * 2);
    w_gwv[l]  = (_Float16*)alloc((size_t)HCH * HCH * 2);
    w_u[l]    = (_Float16*)alloc((size_t)HCH * HCH * 2);
    w_v[l]    = (_Float16*)alloc((size_t)HCH * HCH * 2);
    w_a1[l]   = (_Float16*)alloc((size_t)HCH * 256 * 2);
    w_a2[l]   = (_Float16*)alloc((size_t)384 * HCH * 2);
  }

  auto cvt_w = [&](const float* W, _Float16* dst, int rows, int cols, int colsPad) {
    int total = rows * colsPad;
    cvt_w16_kernel<<<(total + 255) / 256, 256, 0, stream>>>(W, dst, rows, cols, colsPad);
  };
  auto cvt16 = [&](const float* x, _Float16* y, int n) {
    cvt_f16_kernel<<<(n + 255) / 256, 256, 0, stream>>>(x, y, n);
  };
  auto gemm = [&](const _Float16* A, int lda, const _Float16* W, int ldw,
                  const float* bias, float* Cf, _Float16* Ch, int ldc,
                  int M, int Nn, int K, int act) {
    dim3 g((M + 31) / 32, (Nn / 16 + 7) / 8);
    gemm_wmma_kernel<<<g, 256, 0, stream>>>(A, lda, W, ldw, bias, Cf, Ch, ldc, M, Nn, K, act);
  };

  // ---- weight conversions (deterministic, every call) ----
  for (int l = 0; l < NLAYER; ++l) {
    int lb = 5 + l * 26;
    cvt_w(PW(lb + 22), w_phi1[l], HCH, HCH, HCH);
    cvt_w(PW(lb + 24), w_phi2[l], 384, HCH, HCH);
    cvt_w(PW(lb + 4),  w_rbf[l],                 HCH, NRBF, KRBF);  // Ws
    cvt_w(PW(lb + 6),  w_rbf[l] + HCH * KRBF,     HCH, NRBF, KRBF);  // Wv
    cvt_w(PW(lb + 8),  w_rbf[l] + 2 * HCH * KRBF, HCH, NRBF, KRBF);  // Wvs
    cvt_w(PW(lb + 14), w_gwg[l], HCH, 256, 256);
    cvt_w(PW(lb + 16), w_gws[l], HCH, 256, 256);
    cvt_w(PW(lb + 18), w_gwv[l], HCH, HCH, HCH);
    cvt_w(PW(lb + 0),  w_u[l],   HCH, HCH, HCH);
    cvt_w(PW(lb + 2),  w_v[l],   HCH, HCH, HCH);
    cvt_w(PW(lb + 10), w_a1[l],  HCH, 256, 256);
    cvt_w(PW(lb + 12), w_a2[l],  384, HCH, HCH);
  }
  cvt_w(PW(109), w_out1, HCH, HCH, HCH);

  // ---- geometry + embedding init ----
  edge_geom_kernel<<<(E_EDGES + 255) / 256, 256, 0, stream>>>(pos, eidx, ea, rbf16, E_EDGES);
  init_nodes_kernel<<<(int)((3 * NHC + 255) / 256), 256, 0, stream>>>(
      z, embed, s, v, (int)NHC, (int)(3 * NHC));

  const int* rowIdx = eidx;
  const int* colIdx = eidx + E_EDGES;

  for (int l = 0; l < NLAYER; ++l) {
    int lb = 5 + l * 26;
    // snapshots
    cvt16(s, s16, (int)NHC);
    copy_f32_kernel<<<(int)((3 * NHC + 255) / 256), 256, 0, stream>>>(v, v_old, (int)(3 * NHC));
    // fused edge message + scatter (accumulates directly into s, v)
    edge_kernel<<<E_EDGES / TILE_E, 256, 0, stream>>>(
        rowIdx, colIdx, s16, v_old, rbf16, ea,
        w_phi1[l], PW(lb + 23), w_phi2[l], PW(lb + 25),
        w_rbf[l], PW(lb + 5), PW(lb + 7), PW(lb + 9), s, v);

    // ---- gated node update ----
    vnorm_sv_kernel<<<(int)((NHC + 255) / 256), 256, 0, stream>>>(s, v, sv16, N_NODES);
    cvt16(v, v16, (int)(3 * NHC));
    gemm(sv16, 256, w_gwg[l], 256, PW(lb + 15), gpre, nullptr, HCH, N_NODES, HCH, 256, 0);
    gemm(sv16, 256, w_gws[l], 256, PW(lb + 17), gs,   nullptr, HCH, N_NODES, HCH, 256, 0);
    gemm(v16, HCH, w_gwv[l], HCH, PW(lb + 19), wvout, nullptr, HCH, 3 * N_NODES, HCH, HCH, 0);
    gate_apply_kernel<<<N_NODES, 128, 0, stream>>>(
        gpre, gs, PW(lb + 20), PW(lb + 21), wvout, s, v, N_NODES);

    // ---- UV/VV mixing ----
    vnorm_sv_kernel<<<(int)((NHC + 255) / 256), 256, 0, stream>>>(s, v, sv16, N_NODES);
    cvt16(v, v16, (int)(3 * NHC));
    gemm(sv16, 256, w_a1[l], 256, PW(lb + 11), nullptr, t16, HCH, N_NODES, HCH, 256, 1);
    gemm(t16, HCH, w_a2[l], HCH, PW(lb + 13), aout, nullptr, 384, N_NODES, 384, HCH, 0);
    gemm(v16, HCH, w_u[l], HCH, PW(lb + 1), wvout, nullptr, HCH, 3 * N_NODES, HCH, HCH, 0);
    gemm(v16, HCH, w_v[l], HCH, PW(lb + 3), vvout, nullptr, HCH, 3 * N_NODES, HCH, HCH, 0);
    aggregate_kernel<<<(int)((NHC + 255) / 256), 256, 0, stream>>>(
        aout, wvout, vvout, s, v, N_NODES);
  }

  // ---- readout ----
  zero_f32_kernel<<<(NGRAPH * HCH + 255) / 256, 256, 0, stream>>>(sg, NGRAPH * HCH);
  pool_kernel<<<(int)((NHC + 255) / 256), 256, 0, stream>>>(s, batch, sg, N_NODES);
  cvt16(sg, sg16, NGRAPH * HCH);
  gemm(sg16, HCH, w_out1, HCH, PW(110), nullptr, h1_16, HCH, NGRAPH, HCH, HCH, 1);
  out2_kernel<<<NGRAPH, 128, 0, stream>>>(h1_16, PW(111), PW(112), out);
}
